// LinearKAN_35553739276891
// MI455X (gfx1250) — compile-verified, matrix operand order checked by reading, not run on hardware
//
#include <hip/hip_runtime.h>
#include <hip/hip_bf16.h>

// ---------------------------------------------------------------------------
// LinearKAN: out = silu(x) @ w + einsum(basis, c, w)
// Folded into ONE bf16 WMMA GEMM:  A(2048 x 4608) @ WbigT^T(4608 x 512)
//   A[b, k'*512 + i]    = k'==0 ? silu(x[b,i]) : H_{k'-1}(x[b,i])
//   WbigT[o, k'*512+i]  = k'==0 ? w[i,o]       : w[i,o]*c[i,o,k'-1]
// GEMM kernel: double-buffered LDS, async global->LDS DMA (ASYNCcnt),
// K-step 64, 8 x v_wmma_f32_16x16x32_bf16 per wave per stage.
// ---------------------------------------------------------------------------

#define BB   2048
#define II   512
#define OO   512
#define NB   8
#define KK   ((NB + 1) * II)   // 4608
#define KSTEPS (KK / 64)       // 72 stages of K=64

#ifndef __has_builtin
#define __has_builtin(x) 0
#endif

typedef __attribute__((ext_vector_type(4)))  int    v4i;
typedef __attribute__((ext_vector_type(16))) __bf16 v16bf;
typedef __attribute__((ext_vector_type(8)))  float  v8f;

#if defined(__HIP_DEVICE_COMPILE__) && __has_builtin(__builtin_amdgcn_global_load_async_to_lds_b128)
#define KAN_ASYNC 1
typedef __attribute__((address_space(1))) v4i GQuad;   // global int4
typedef __attribute__((address_space(3))) v4i LQuad;   // LDS int4
#else
#define KAN_ASYNC 0
#endif

union Frag {
    uint4  q[2];
    v16bf  v;
};

static __device__ __forceinline__ unsigned short f32_to_bf16(float f) {
    unsigned int u = __float_as_uint(f);
    unsigned int r = u + 0x7FFFu + ((u >> 16) & 1u);   // round-to-nearest-even
    return (unsigned short)(r >> 16);
}

static __device__ __forceinline__ void wait_async0() {
#if KAN_ASYNC
#if __has_builtin(__builtin_amdgcn_s_wait_asynccnt)
    __builtin_amdgcn_s_wait_asynccnt(0);
#else
    asm volatile("s_wait_asynccnt 0x0" ::: "memory");
#endif
#endif
}

// ---------------------------------------------------------------------------
// Kernel 1: activation matrix A (bf16, 2048 x 4608)
// ---------------------------------------------------------------------------
__global__ __launch_bounds__(256) void kan_prep_act(const float* __restrict__ x,
                                                    unsigned short* __restrict__ A) {
    int gid = blockIdx.x * blockDim.x + threadIdx.x;   // 0 .. B*I-1
    int b = gid >> 9;
    int i = gid & 511;
    float v = x[gid];
    float s = v / (1.0f + __expf(-v));                 // silu

    float h[NB];
    h[0] = 1.0f;
    h[1] = 2.0f * v;
#pragma unroll
    for (int k = 1; k < NB - 1; ++k)
        h[k + 1] = 2.0f * v * h[k] - 2.0f * (float)k * h[k - 1];

    size_t base = (size_t)b * KK + i;
    A[base] = f32_to_bf16(s);
#pragma unroll
    for (int k = 0; k < NB; ++k)
        A[base + (size_t)(k + 1) * II] = f32_to_bf16(h[k]);
}

// ---------------------------------------------------------------------------
// Kernel 2: fused weights WbigT (bf16, 512(N) x 4608(K), row-major)
// ---------------------------------------------------------------------------
__global__ __launch_bounds__(256) void kan_prep_w(const float* __restrict__ w,
                                                  const float* __restrict__ c,
                                                  unsigned short* __restrict__ Wt) {
    int gid = blockIdx.x * blockDim.x + threadIdx.x;   // 0 .. I*O-1
    int o = gid >> 9;
    int i = gid & 511;
    float wv = w[(size_t)i * OO + o];
    const float* cp = c + ((size_t)i * OO + o) * NB;

    size_t base = (size_t)o * KK + i;
    Wt[base] = f32_to_bf16(wv);
#pragma unroll
    for (int k = 0; k < NB; ++k)
        Wt[base + (size_t)(k + 1) * II] = f32_to_bf16(wv * cp[k]);
}

// ---------------------------------------------------------------------------
// Kernel 3: bf16 WMMA GEMM, double-buffered async LDS pipeline.
// Block = 128 thr (4 waves), 64x64 output tile, wave = 32x32 = 2x2 frags.
// ---------------------------------------------------------------------------
__global__ __launch_bounds__(128) void kan_wmma_gemm(const unsigned short* __restrict__ A,
                                                     const unsigned short* __restrict__ Bt,
                                                     float* __restrict__ out) {
    __shared__ unsigned short As[2][64 * 64];   // 2 x 8 KB
    __shared__ unsigned short Bs[2][64 * 64];   // 2 x 8 KB

    const int m0   = blockIdx.y * 64;
    const int n0   = blockIdx.x * 64;
    const int tid  = threadIdx.x;
    const int lane = tid & 31;
    const int wid  = tid >> 5;
    const int wm   = (wid >> 1) * 32;
    const int wn   = (wid & 1) * 32;
    const int lrow  = lane & 15;
    const int khalf = (lane >> 4) * 8;   // lanes 0-15: K 0-7/16-23; 16-31: K 8-15/24-31

    // ---- stage fill: 64 rows x 64 cols bf16 = 512 uint4 per matrix ----
    auto fill = [&](int stage, int k0) {
#pragma unroll
        for (int t = tid; t < 512; t += 128) {
            int row = t >> 3, c8 = t & 7;
            const unsigned short* ga = A  + (size_t)(m0 + row) * KK + k0 + c8 * 8;
            const unsigned short* gb = Bt + (size_t)(n0 + row) * KK + k0 + c8 * 8;
#if KAN_ASYNC
            __builtin_amdgcn_global_load_async_to_lds_b128(
                (GQuad*)ga, (LQuad*)(As[stage] + t * 8), 0, 0);
            __builtin_amdgcn_global_load_async_to_lds_b128(
                (GQuad*)gb, (LQuad*)(Bs[stage] + t * 8), 0, 0);
#else
            ((uint4*)As[stage])[t] = *(const uint4*)ga;
            ((uint4*)Bs[stage])[t] = *(const uint4*)gb;
#endif
        }
    };

    v8f acc[2][2];
#pragma unroll
    for (int a = 0; a < 2; ++a)
#pragma unroll
        for (int b = 0; b < 2; ++b)
            acc[a][b] = (v8f){0.f, 0.f, 0.f, 0.f, 0.f, 0.f, 0.f, 0.f};

    fill(0, 0);                                   // prefill stage 0

    for (int kt = 0; kt < KSTEPS; ++kt) {
        const int cur = kt & 1;

        wait_async0();                            // stage kt resident in LDS
        __syncthreads();                          // visible to all; prior readers done

        if (kt + 1 < KSTEPS)
            fill(1 - cur, (kt + 1) * 64);         // async fill overlaps compute

        // light prefetch of the stage after next into L2
        if (kt + 2 < KSTEPS) {
            int r = tid & 63;
            __builtin_prefetch(A  + (size_t)(m0 + r) * KK + (kt + 2) * 64, 0, 1);
            __builtin_prefetch(Bt + (size_t)(n0 + r) * KK + (kt + 2) * 64, 0, 1);
        }

        // ---- 2 K-slices of 32, 4 WMMAs each ----
#pragma unroll
        for (int kc = 0; kc < 64; kc += 32) {
            Frag fa[2], fb[2];
#pragma unroll
            for (int s = 0; s < 2; ++s) {
                const uint4* pa = (const uint4*)(As[cur] + (wm + s * 16 + lrow) * 64 + kc + khalf);
                fa[s].q[0] = pa[0];               // K kc+khalf   .. +7
                fa[s].q[1] = pa[2];               // K kc+khalf+16.. +23
                const uint4* pb = (const uint4*)(Bs[cur] + (wn + s * 16 + lrow) * 64 + kc + khalf);
                fb[s].q[0] = pb[0];
                fb[s].q[1] = pb[2];
            }
#pragma unroll
            for (int sm = 0; sm < 2; ++sm)
#pragma unroll
                for (int sn = 0; sn < 2; ++sn)
                    acc[sm][sn] = __builtin_amdgcn_wmma_f32_16x16x32_bf16(
                        false, fa[sm].v, false, fb[sn].v,
                        (short)0, acc[sm][sn], false, false);
        }
        __syncthreads();                          // compute done before buffer reuse
    }

    // ---- epilogue: VGPR r holds M = r + 8*(lane>=16), N = lane&15 ----
#pragma unroll
    for (int sm = 0; sm < 2; ++sm)
#pragma unroll
        for (int sn = 0; sn < 2; ++sn) {
            int col   = n0 + wn + sn * 16 + lrow;
            int rbase = m0 + wm + sm * 16 + (lane >> 4) * 8;
#pragma unroll
            for (int r = 0; r < 8; ++r)
                out[(size_t)(rbase + r) * OO + col] = acc[sm][sn][r];
        }
}

// ---------------------------------------------------------------------------
extern "C" void kernel_launch(void* const* d_in, const int* in_sizes, int n_in,
                              void* d_out, int out_size, void* d_ws, size_t ws_size,
                              hipStream_t stream) {
    const float* x = (const float*)d_in[0];   // (B, I)
    const float* w = (const float*)d_in[1];   // (I, O)
    const float* c = (const float*)d_in[2];   // (I, O, n)
    float* out = (float*)d_out;               // (B, O) f32

    unsigned short* Aflat = (unsigned short*)d_ws;                       // 2048*4608 bf16
    unsigned short* Wt    = (unsigned short*)((char*)d_ws +
                             (size_t)BB * KK * sizeof(unsigned short));  // 512*4608 bf16

    kan_prep_act<<<(BB * II) / 256, 256, 0, stream>>>(x, Aflat);
    kan_prep_w<<<(II * OO) / 256, 256, 0, stream>>>(w, c, Wt);

    dim3 grid(OO / 64, BB / 64);   // 8 x 32 = 256 workgroups
    kan_wmma_gemm<<<grid, 128, 0, stream>>>(Aflat, Wt, out);
}
